// RULSTM_44504451121421
// MI455X (gfx1250) — matrix operand order repaired and checked by reading, not compiled, not run on hardware
//
#include <hip/hip_runtime.h>
#include <hip/hip_bf16.h>

// ---------------------------------------------------------------------------
// Problem constants (from reference)
// ---------------------------------------------------------------------------
#define S_   16
#define B_   512
#define H_   1024
#define A_   2513
#define FIN_ 2048
#define G_   (4 * H_)   // 4096 gate width

// ---------------------------------------------------------------------------
// Vector types for WMMA
// ---------------------------------------------------------------------------
typedef __attribute__((ext_vector_type(16))) __bf16 v16bf;
typedef __attribute__((ext_vector_type(8)))  __bf16 v8bf;
typedef __attribute__((ext_vector_type(8)))  float  v8f;
typedef __attribute__((ext_vector_type(4)))  unsigned int v4u;

union V16U { v16bf v; v8bf h[2]; };

// ---------------------------------------------------------------------------
// GEMM: C[M,N] = A[M,K] (bf16, row major) x Bw[N,K]^T (bf16, row major)
//               (+ bias1[n] + bias2[n]) (+ Cinit[M,N] when HAS_CINIT)
// Tile: 128x128x32, 256 threads = 8 waves (2 M-waves x 4 N-waves),
// each wave owns a 64x32 sub-tile = 4x2 WMMA 16x16 accumulators.
//
// Tile feed uses CDNA5 async global->LDS DMA (GLOBAL_LOAD_ASYNC_TO_LDS_B128,
// ASYNCcnt) with a ping-pong LDS double buffer so the k+1 tile streams in
// while the k tile runs its 8 WMMAs. Every thread issues exactly 4 async ops
// per K-step (OOB weight rows are clamped, then zero-patched after the wait),
// so `s_wait_asynccnt 4` deterministically retires the oldest tile.
//
// M, K must be multiples of 128 / 32 (true for all calls); N is bounds-checked.
// ---------------------------------------------------------------------------
#define BM 128
#define BN 128
#define BK 32
#define LDT 40   // LDS row stride in bf16 halves (80B: padded, 16B aligned)

template <bool OUT_BF16, bool HAS_CINIT>
__global__ __launch_bounds__(256)
void gemm_bf16_wmma(const __bf16* __restrict__ A, int lda,
                    const __bf16* __restrict__ Bw, int ldb,
                    void* __restrict__ Cout, int ldc,
                    const float* __restrict__ bias1,
                    const float* __restrict__ bias2,
                    const float* __restrict__ Cinit, int ldci,
                    int M, int N, int K)
{
    __shared__ __align__(16) __bf16 As[2][BM * LDT];
    __shared__ __align__(16) __bf16 Bs[2][BN * LDT];

    const int tid   = threadIdx.x;
    const int lane  = tid & 31;
    const int wid   = tid >> 5;
    const int waveM = wid >> 2;        // 0..1
    const int waveN = wid & 3;         // 0..3
    const int m0    = blockIdx.y * BM;
    const int n0    = blockIdx.x * BN;

    const int laneMod  = lane & 15;    // row/col within 16
    const int laneHalf = lane >> 4;    // 0 or 1 (K-half / M-half selector)

    v8f acc[4][2];
    #pragma unroll
    for (int mi = 0; mi < 4; ++mi)
        #pragma unroll
        for (int ni = 0; ni < 2; ++ni) {
            if (HAS_CINIT) {
                const int ncol = n0 + waveN * 32 + ni * 16 + laneMod;
                const int mrow = m0 + waveM * 64 + mi * 16 + laneHalf * 8;
                #pragma unroll
                for (int r = 0; r < 8; ++r)
                    acc[mi][ni][r] = (ncol < N)
                        ? Cinit[(size_t)(mrow + r) * ldci + ncol] : 0.0f;
            } else {
                v8f z = {};
                acc[mi][ni] = z;
            }
        }

    // cooperative tile loads: thread -> (row, 32B segment)
    const int ldRow = tid >> 1;          // 0..127
    const int ldSeg = (tid & 1) * 16;    // 0 or 16 halves (32B each)

    const int  nrow  = n0 + ldRow;
    const bool bIn   = nrow < N;
    const int  nrowc = bIn ? nrow : (N - 1);      // clamp: always issue loads
    const unsigned segB = (unsigned)((ldRow * LDT + ldSeg) * 2);  // LDS bytes
    const unsigned asBase0 = (unsigned)(uintptr_t)&As[0][0];
    const unsigned asBase1 = (unsigned)(uintptr_t)&As[1][0];
    const unsigned bsBase0 = (unsigned)(uintptr_t)&Bs[0][0];
    const unsigned bsBase1 = (unsigned)(uintptr_t)&Bs[1][0];

    // Issue the 4 async global->LDS b128 transfers for one K-tile.
    auto issueTile = [&](int buf, int k0) {
        const __bf16* srcA = A + (size_t)(m0 + ldRow) * lda + k0 + ldSeg;
        const __bf16* srcB = Bw + (size_t)nrowc * ldb + k0 + ldSeg;
        const unsigned dA = (buf ? asBase1 : asBase0) + segB;
        const unsigned dB = (buf ? bsBase1 : bsBase0) + segB;
        asm volatile("global_load_async_to_lds_b128 %0, %1, off"
                     :: "v"(dA), "v"((const void*)srcA) : "memory");
        asm volatile("global_load_async_to_lds_b128 %0, %1, off"
                     :: "v"(dA + 16u), "v"((const void*)(srcA + 8)) : "memory");
        asm volatile("global_load_async_to_lds_b128 %0, %1, off"
                     :: "v"(dB), "v"((const void*)srcB) : "memory");
        asm volatile("global_load_async_to_lds_b128 %0, %1, off"
                     :: "v"(dB + 16u), "v"((const void*)(srcB + 8)) : "memory");
    };

    const int nK = K / BK;
    issueTile(0, 0);

    for (int kt = 0; kt < nK; ++kt) {
        const int buf = kt & 1;

        if (kt + 1 < nK) {
            issueTile(buf ^ 1, (kt + 1) * BK);
            // oldest 4 (current tile) retired; next tile's 4 may stay in flight
            asm volatile("s_wait_asynccnt 0x4" ::: "memory");
        } else {
            asm volatile("s_wait_asynccnt 0x0" ::: "memory");
        }

        // zero-patch OOB weight rows of the *current* buffer (after the wait,
        // before the barrier) so the WMMA sees zeros, not clamped garbage
        if (!bIn) {
            v4u z = {};
            v4u* d = (v4u*)&Bs[buf][ldRow * LDT + ldSeg];
            d[0] = z; d[1] = z;
        }
        __syncthreads();

        // Assemble WMMA fragments per the CDNA5 bf16 layouts:
        //  A 16x32: lane<16 holds row M=lane, K {0..7,16..23};
        //           lane>=16 holds row M=lane-16, K {8..15,24..31}
        //  B 32x16: lane<16 holds col N=lane, K {0..7,16..23};
        //           lane>=16 holds col N=lane-16, K {8..15,24..31}
        V16U afr[4], bfr[2];
        #pragma unroll
        for (int mi = 0; mi < 4; ++mi) {
            const int r  = waveM * 64 + mi * 16 + laneMod;
            const int kh = laneHalf * 8;
            afr[mi].h[0] = *(const v8bf*)&As[buf][r * LDT + kh];
            afr[mi].h[1] = *(const v8bf*)&As[buf][r * LDT + kh + 16];
        }
        #pragma unroll
        for (int ni = 0; ni < 2; ++ni) {
            const int r  = waveN * 32 + ni * 16 + laneMod;
            const int kh = laneHalf * 8;
            bfr[ni].h[0] = *(const v8bf*)&Bs[buf][r * LDT + kh];
            bfr[ni].h[1] = *(const v8bf*)&Bs[buf][r * LDT + kh + 16];
        }

        #pragma unroll
        for (int mi = 0; mi < 4; ++mi)
            #pragma unroll
            for (int ni = 0; ni < 2; ++ni)
                acc[mi][ni] = __builtin_amdgcn_wmma_f32_16x16x32_bf16(
                    false, afr[mi].v, false, bfr[ni].v,
                    (short)0, acc[mi][ni], false, false);

        __syncthreads();   // all waves done reading buf before it is re-filled
    }

    // epilogue: C/D layout -> lane (n, m-half), VGPR r = m within half
    #pragma unroll
    for (int mi = 0; mi < 4; ++mi)
        #pragma unroll
        for (int ni = 0; ni < 2; ++ni) {
            const int ncol = n0 + waveN * 32 + ni * 16 + laneMod;
            if (ncol >= N) continue;
            float bv = 0.0f;
            if (bias1) bv += bias1[ncol];
            if (bias2) bv += bias2[ncol];
            const int mrow = m0 + waveM * 64 + mi * 16 + laneHalf * 8;
            #pragma unroll
            for (int r = 0; r < 8; ++r) {
                const float v = acc[mi][ni][r] + bv;
                if (OUT_BF16)
                    ((__bf16*)Cout)[(size_t)(mrow + r) * ldc + ncol] = (__bf16)v;
                else
                    ((float*)Cout)[(size_t)(mrow + r) * ldc + ncol] = v;
            }
        }
}

// ---------------------------------------------------------------------------
// Elementwise kernels
// ---------------------------------------------------------------------------
__global__ void k_cvt_bf16(__bf16* __restrict__ dst, const float* __restrict__ src, size_t n)
{
    for (size_t i = (size_t)blockIdx.x * blockDim.x + threadIdx.x; i < n;
         i += (size_t)gridDim.x * blockDim.x)
        dst[i] = (__bf16)src[i];
}

// fold concat([x,x]) @ Wih^T into x @ (Wih[:, :H] + Wih[:, H:])^T
__global__ void k_fold_ih(__bf16* __restrict__ dst, const float* __restrict__ src,
                          int Gd, int Hd)
{
    const size_t n = (size_t)Gd * Hd;
    for (size_t i = (size_t)blockIdx.x * blockDim.x + threadIdx.x; i < n;
         i += (size_t)gridDim.x * blockDim.x) {
        const size_t g = i / Hd, k = i % Hd;
        dst[i] = (__bf16)(src[g * (2 * Hd) + k] + src[g * (2 * Hd) + Hd + k]);
    }
}

__global__ void k_zero_state(float* __restrict__ c, __bf16* __restrict__ h, size_t n)
{
    for (size_t i = (size_t)blockIdx.x * blockDim.x + threadIdx.x; i < n;
         i += (size_t)gridDim.x * blockDim.x) {
        c[i] = 0.0f;
        h[i] = (__bf16)0.0f;
    }
}

__device__ __forceinline__ float sigf(float x) { return 1.0f / (1.0f + __expf(-x)); }

// gates layout [B, 4H] with PyTorch order i,f,g,o; updates c (f32), h (bf16)
__global__ void k_lstm_cell(const float* __restrict__ gates,
                            float* __restrict__ c, __bf16* __restrict__ h,
                            int Hd, int Gd, size_t n)
{
    for (size_t i = (size_t)blockIdx.x * blockDim.x + threadIdx.x; i < n;
         i += (size_t)gridDim.x * blockDim.x) {
        const size_t b = i / Hd, j = i % Hd;
        const size_t base = b * Gd;
        const float ig = gates[base + j];
        const float fg = gates[base + Hd + j];
        const float gg = gates[base + 2 * Hd + j];
        const float og = gates[base + 3 * Hd + j];
        const float cn = sigf(fg) * c[i] + sigf(ig) * tanhf(gg);
        c[i] = cn;
        h[i] = (__bf16)(sigf(og) * tanhf(cn));
    }
}

// ---------------------------------------------------------------------------
// Host orchestration
// ---------------------------------------------------------------------------
extern "C" void kernel_launch(void* const* d_in, const int* in_sizes, int n_in,
                              void* d_out, int out_size, void* d_ws, size_t ws_size,
                              hipStream_t stream)
{
    (void)in_sizes; (void)n_in; (void)out_size; (void)ws_size;

    const float* inputs = (const float*)d_in[0];
    const float* W1     = (const float*)d_in[1];
    const float* b1     = (const float*)d_in[2];
    const float* Wih_r  = (const float*)d_in[3];
    const float* Whh_r  = (const float*)d_in[4];
    const float* bih_r  = (const float*)d_in[5];
    const float* bhh_r  = (const float*)d_in[6];
    const float* Wih_u  = (const float*)d_in[7];
    const float* Whh_u  = (const float*)d_in[8];
    const float* bih_u  = (const float*)d_in[9];
    const float* bhh_u  = (const float*)d_in[10];
    const float* Wc     = (const float*)d_in[11];
    const float* bc     = (const float*)d_in[12];
    float* out = (float*)d_out;

    // workspace carve-out
    char* p = (char*)d_ws;
    auto alloc = [&](size_t bytes) -> void* {
        void* r = (void*)p;
        p += (bytes + 255) & ~(size_t)255;
        return r;
    };
    const size_t nInput = (size_t)S_ * B_ * FIN_;
    const size_t nX     = (size_t)S_ * B_ * H_;
    const size_t nGH    = (size_t)G_ * H_;
    const size_t nBH    = (size_t)B_ * H_;
    const size_t nBG    = (size_t)B_ * G_;

    __bf16* in_b  = (__bf16*)alloc(nInput * 2);
    __bf16* x_b   = (__bf16*)alloc(nX * 2);
    __bf16* W1b   = (__bf16*)alloc((size_t)H_ * FIN_ * 2);
    __bf16* Wr_ih = (__bf16*)alloc(nGH * 2);   // folded
    __bf16* Wr_hh = (__bf16*)alloc(nGH * 2);
    __bf16* Wu_ih = (__bf16*)alloc(nGH * 2);   // folded
    __bf16* Wu_hh = (__bf16*)alloc(nGH * 2);
    __bf16* Wcb   = (__bf16*)alloc((size_t)A_ * H_ * 2);
    float*  xg_r  = (float*)alloc((size_t)S_ * B_ * G_ * 4);
    float*  xg_u  = (float*)alloc(nBG * 4);
    float*  gates = (float*)alloc(nBG * 4);
    float*  c_st  = (float*)alloc(nBH * 4);
    __bf16* h_b   = (__bf16*)alloc(nBH * 2);

    // --- convert / fold weights & inputs to bf16 ---
    k_cvt_bf16<<<2048, 256, 0, stream>>>(in_b, inputs, nInput);
    k_cvt_bf16<<<512,  256, 0, stream>>>(W1b, W1, (size_t)H_ * FIN_);
    k_fold_ih<<<1024,  256, 0, stream>>>(Wr_ih, Wih_r, G_, H_);
    k_cvt_bf16<<<1024, 256, 0, stream>>>(Wr_hh, Whh_r, nGH);
    k_fold_ih<<<1024,  256, 0, stream>>>(Wu_ih, Wih_u, G_, H_);
    k_cvt_bf16<<<1024, 256, 0, stream>>>(Wu_hh, Whh_u, nGH);
    k_cvt_bf16<<<1024, 256, 0, stream>>>(Wcb, Wc, (size_t)A_ * H_);
    k_zero_state<<<512, 256, 0, stream>>>(c_st, h_b, nBH);

    const int SB = S_ * B_;   // 8192

    // --- fc1: x = inputs @ W1^T + b1  -> bf16 [S*B, H] ---
    gemm_bf16_wmma<true, false><<<dim3(H_ / BN, SB / BM), 256, 0, stream>>>(
        in_b, FIN_, W1b, FIN_, (void*)x_b, H_,
        b1, nullptr, nullptr, 0, SB, H_, FIN_);

    // --- xg_r = x @ Wr_ih^T + (bih_r + bhh_r)  -> f32 [S*B, 4H] ---
    gemm_bf16_wmma<false, false><<<dim3(G_ / BN, SB / BM), 256, 0, stream>>>(
        x_b, H_, Wr_ih, H_, (void*)xg_r, G_,
        bih_r, bhh_r, nullptr, 0, SB, G_, H_);

    // --- xg_u[15] = x[15] @ Wu_ih^T + (bih_u + bhh_u)  -> f32 [B, 4H] ---
    // Only t = S-1 feeds the output (y[:, -1] only uses h_n[S-1]).
    gemm_bf16_wmma<false, false><<<dim3(G_ / BN, B_ / BM), 256, 0, stream>>>(
        x_b + (size_t)(S_ - 1) * B_ * H_, H_, Wu_ih, H_, (void*)xg_u, G_,
        bih_u, bhh_u, nullptr, 0, B_, G_, H_);

    // --- rolling LSTM: 16 steps -> (hs[15], cs[15]) in (h_b, c_st) ---
    for (int t = 0; t < S_; ++t) {
        gemm_bf16_wmma<false, true><<<dim3(G_ / BN, B_ / BM), 256, 0, stream>>>(
            h_b, H_, Wr_hh, H_, (void*)gates, G_,
            nullptr, nullptr, xg_r + (size_t)t * B_ * G_, G_, B_, G_, H_);
        k_lstm_cell<<<1024, 256, 0, stream>>>(gates, c_st, h_b, H_, G_, nBH);
    }

    // --- unrolling LSTM at t = 15: exactly 2 steps ---
    for (int it = 0; it < 2; ++it) {
        gemm_bf16_wmma<false, true><<<dim3(G_ / BN, B_ / BM), 256, 0, stream>>>(
            h_b, H_, Wu_hh, H_, (void*)gates, G_,
            nullptr, nullptr, xg_u, G_, B_, G_, H_);
        k_lstm_cell<<<1024, 256, 0, stream>>>(gates, c_st, h_b, H_, G_, nBH);
    }

    // --- classifier: y = h @ Wc^T + bc -> f32 [B, A] ---
    gemm_bf16_wmma<false, false><<<dim3((A_ + BN - 1) / BN, B_ / BM), 256, 0, stream>>>(
        h_b, H_, Wcb, H_, (void*)out, A_,
        bc, nullptr, nullptr, 0, B_, A_, H_);
}
